// Model_6640019440208
// MI455X (gfx1250) — matrix operand order repaired and checked by reading, not compiled, not run on hardware
//
#include <hip/hip_runtime.h>

typedef __attribute__((ext_vector_type(16))) _Float16 v16h;
typedef __attribute__((ext_vector_type(8)))  _Float16 v8h;
typedef __attribute__((ext_vector_type(8)))  float    v8f;

__device__ __forceinline__ float lrelu_f(float v) { return v >= 0.f ? v : 0.2f * v; }
__device__ __forceinline__ int   imin(int a, int b) { return a < b ? a : b; }
__device__ __forceinline__ int   imax(int a, int b) { return a > b ? a : b; }

// ---------------------------------------------------------------------------
// Implicit-im2col 3x3 conv as f16 WMMA GEMM (16x16x32, fp32 accumulate).
// Padded-K layout: each input channel occupies 16 K-slots (taps 0..8 real,
// 9..15 zero), Kp = roundup(Cin*16, 32). One K-chunk of 32 = two channels:
// each lane's 16-element B run is ONE channel (ch = 2c + half) with
// compile-time tap indices.
// Out-of-bounds handling is via ADDRESS selection into a zeroed page
// (v_cndmask on the offset), so every gather is an unconditional
// global_load_u16 -> no exec-mask branches, no per-load waits.
// MT (1..4) cout sub-tiles of 16 share each B fragment (template parameter
// so accumulators stay pinned).
// ---------------------------------------------------------------------------
template <int MT>
__global__ __launch_bounds__(32) void conv3x3_wmma_f16(
    const _Float16* __restrict__ act, const _Float16* __restrict__ wh,
    const float* __restrict__ bias, const _Float16* __restrict__ zpage,
    float* __restrict__ out,
    int Cin, int Hin, int Win, int Cout, int Hout, int Wout,
    int stride, int relu, int Kp, int mbase0)
{
    const int lane = threadIdx.x;
    const int half = lane >> 4;
    const int lm   = lane & 15;

    const int WT = (Wout + 15) >> 4;
    int t = blockIdx.x;
    const int wt = t % WT; t /= WT;
    const int oy = t % Hout;
    const int bb = t / Hout;
    const int mbase = mbase0 + blockIdx.y * (MT << 4);

    const int ox  = (wt << 4) + lm;
    const bool oxv = ox < Wout;
    const int iy0 = oy * stride - 1;
    const int ix0 = ox * stride - 1;
    const int HW  = Hin * Win;
    const _Float16* actB = act + (size_t)bb * Cin * HW;
    const int zBase = (int)(zpage - actB);   // offset of zero page from actB

    // 9 precomputed clamped tap offsets + validity (oxv folded in)
    int  offs[9];
    bool mv[9];
#pragma unroll
    for (int tap = 0; tap < 9; ++tap) {
        int ky = tap / 3, kx = tap - 3 * (tap / 3);   // compile-time
        int iy = iy0 + ky, ix = ix0 + kx;
        mv[tap] = oxv && ((unsigned)iy < (unsigned)Hin)
                      && ((unsigned)ix < (unsigned)Win);
        int iyc = imin(imax(iy, 0), Hin - 1);
        int ixc = imin(imax(ix, 0), Win - 1);
        offs[tap] = iyc * Win + ixc;
    }

    // per-sub-tile weight row pointers (row clamped; garbage rows never stored)
    const _Float16* wrow[MT];
#pragma unroll
    for (int mt = 0; mt < MT; ++mt) {
        int mc = imin(mbase + (mt << 4) + lm, Cout - 1);
        wrow[mt] = wh + (size_t)mc * Kp + (half << 4);
    }

    v8f acc[MT];
#pragma unroll
    for (int i = 0; i < MT; ++i) acc[i] = (v8f){0.f,0.f,0.f,0.f,0.f,0.f,0.f,0.f};

    const int nChunks = Kp >> 5;
    for (int c = 0; c < nChunks; ++c) {
        // ---- B fragment: this lane's channel = 2c + half ----
        const int ch     = (c << 1) + half;
        const bool cv    = ch < Cin;
        const int chBase = imin(ch, Cin - 1) * HW;
        const _Float16* ap = actB + chBase;
        const int zo     = zBase - chBase;          // ap[zo] == zero page
        union { v16h v; _Float16 e[16]; } bu;
#pragma unroll
        for (int tap = 0; tap < 9; ++tap) {
            int o = (cv && mv[tap]) ? offs[tap] : zo;
            bu.e[tap] = ap[o];                      // always-safe address
        }
#pragma unroll
        for (int tap = 9; tap < 16; ++tap) bu.e[tap] = (_Float16)0.f;

        // ---- A fragments (two aligned 16B loads each), then MT wmma ----
        v16h a[MT];
#pragma unroll
        for (int mt = 0; mt < MT; ++mt) {
            const _Float16* wr = wrow[mt] + (c << 5);
            union { v16h v; v8h h[2]; } au;
            au.h[0] = *(const v8h*)(wr);
            au.h[1] = *(const v8h*)(wr + 8);
            a[mt] = au.v;
        }
#pragma unroll
        for (int mt = 0; mt < MT; ++mt)
            acc[mt] = __builtin_amdgcn_wmma_f32_16x16x32_f16(
                false, a[mt], false, bu.v, (short)0, acc[mt], false, false);
    }

    if (oxv) {
#pragma unroll
        for (int mt = 0; mt < MT; ++mt) {
#pragma unroll
            for (int j = 0; j < 8; ++j) {
                int co = mbase + (mt << 4) + (half << 3) + j;
                if (co < Cout) {
                    float v = acc[mt][j] + bias[co];
                    if (relu) v = lrelu_f(v);
                    out[(((size_t)bb * Cout + co) * Hout + oy) * Wout + ox] = v;
                }
            }
        }
    }
}

// ---------------------------------------------------------------------------
// Precision staging kernels
// ---------------------------------------------------------------------------
__global__ void zfill_kernel(_Float16* __restrict__ o, int total)
{
    int i = blockIdx.x * blockDim.x + threadIdx.x;
    if (i < total) o[i] = (_Float16)0.f;
}

// fp32 OIHW weights (Cout x Cin*9) -> f16 padded layout (Cout x Kp),
// channel ci at Kp-offset ci*16, taps 9..15 and channels >= Cin are zero.
__global__ void wcvt_kernel(const float* __restrict__ w, _Float16* __restrict__ o,
                            int Cin, int Kp, int total)
{
    int i = blockIdx.x * blockDim.x + threadIdx.x;
    if (i >= total) return;
    int co = i / Kp, r = i - co * Kp;
    int ci = r >> 4, tap = r & 15;
    float v = (tap < 9 && ci < Cin) ? w[((size_t)co * Cin + ci) * 9 + tap] : 0.f;
    o[i] = (_Float16)v;
}

__global__ void acvt_kernel(const float* __restrict__ in, _Float16* __restrict__ o, int total)
{
    int i = blockIdx.x * blockDim.x + threadIdx.x;
    if (i >= total) return;
    o[i] = (_Float16)in[i];
}

// ---------------------------------------------------------------------------
// Elementwise helpers (fp32 master activations)
// ---------------------------------------------------------------------------
__global__ void submean_kernel(const float* __restrict__ in, float* __restrict__ out,
                               int HW, int total, float m0, float m1, float m2)
{
    int i = blockIdx.x * blockDim.x + threadIdx.x;
    if (i >= total) return;
    int c = (i / HW) % 3;
    float mm = (c == 0) ? m0 : ((c == 1) ? m1 : m2);
    out[i] = in[i] - mm;
}

// cost volume, _IDX permutation folded: out ch j <-> shift (4 - j%9, 4 - j/9)
__global__ void corr_kernel(const float* __restrict__ f1, const float* __restrict__ f2,
                            float* __restrict__ out, int C, int Hc, int Wc,
                            int Ctot, float invC, int total)
{
    int i = blockIdx.x * blockDim.x + threadIdx.x;
    if (i >= total) return;
    int x = i % Wc; int t = i / Wc;
    int y = t % Hc; t /= Hc;
    int j = t % 81; int b = t / 81;
    int sy = y + 4 - (j % 9);
    int sx = x + 4 - (j / 9);
    float s = 0.f;
    if ((unsigned)sy < (unsigned)Hc && (unsigned)sx < (unsigned)Wc) {
        const size_t hw = (size_t)Hc * Wc;
        const float* p1 = f1 + (size_t)b * C * hw + (size_t)y * Wc + x;
        const float* p2 = f2 + (size_t)b * C * hw + (size_t)sy * Wc + sx;
        for (int c = 0; c < C; ++c) s += p1[c * hw] * p2[c * hw];
    }
    out[(((size_t)b * Ctot + j) * Hc + y) * Wc + x] = s * invC;
}

__global__ void warp_kernel(const float* __restrict__ img, const float* __restrict__ flo,
                            float scale, float* __restrict__ out,
                            int C, int Hc, int Wc, int total)
{
    int i = blockIdx.x * blockDim.x + threadIdx.x;
    if (i >= total) return;
    int xp = i % Wc; int t = i / Wc;
    int y = t % Hc; t /= Hc;
    int c = t % C; int b = t / C;
    size_t fb = (size_t)b * 2 * Hc * Wc;
    float gx = flo[fb + (size_t)y * Wc + xp] * scale + (float)xp;
    float gy = flo[fb + ((size_t)Hc + y) * Wc + xp] * scale + (float)y;
    gx = fminf(fmaxf(gx, 0.f), (float)(Wc - 1));
    gy = fminf(fmaxf(gy, 0.f), (float)(Hc - 1));
    int x0 = (int)floorf(gx), y0 = (int)floorf(gy);
    float wx = gx - (float)x0, wy = gy - (float)y0;
    int x1 = imin(x0 + 1, Wc - 1), y1 = imin(y0 + 1, Hc - 1);
    const float* p = img + ((size_t)b * C + c) * Hc * Wc;
    float v00 = p[(size_t)y0 * Wc + x0], v01 = p[(size_t)y0 * Wc + x1];
    float v10 = p[(size_t)y1 * Wc + x0], v11 = p[(size_t)y1 * Wc + x1];
    out[i] = v00 * (1.f - wy) * (1.f - wx) + v01 * (1.f - wy) * wx
           + v10 * wy * (1.f - wx) + v11 * wy * wx;
}

__global__ void upsample2_kernel(const float* __restrict__ in, float* __restrict__ out,
                                 int C, int Hin, int Win, float outScale, int total)
{
    int Hout = 2 * Hin, Wout = 2 * Win;
    int i = blockIdx.x * blockDim.x + threadIdx.x;
    if (i >= total) return;
    int x = i % Wout; int t = i / Wout;
    int y = t % Hout; t /= Hout;
    int c = t % C; int b = t / C;
    float ry = (float)(Hin - 1) / (float)(Hout - 1);
    float rx = (float)(Win - 1) / (float)(Wout - 1);
    float sy = y * ry, sx = x * rx;
    int y0 = (int)sy, x0 = (int)sx;
    float wy = sy - (float)y0, wx = sx - (float)x0;
    int y1 = imin(y0 + 1, Hin - 1), x1 = imin(x0 + 1, Win - 1);
    const float* p = in + ((size_t)b * C + c) * Hin * Win;
    float v = p[(size_t)y0 * Win + x0] * (1.f - wy) * (1.f - wx)
            + p[(size_t)y0 * Win + x1] * (1.f - wy) * wx
            + p[(size_t)y1 * Win + x0] * wy * (1.f - wx)
            + p[(size_t)y1 * Win + x1] * wy * wx;
    out[i] = v * outScale;
}

__global__ void copycat_kernel(const float* __restrict__ src, float* __restrict__ dst,
                               int C, int HW, int Ctot, int coff, int total)
{
    int i = blockIdx.x * blockDim.x + threadIdx.x;
    if (i >= total) return;
    int p = i % HW; int t = i / HW;
    int c = t % C; int b = t / C;
    dst[((size_t)b * Ctot + coff + c) * HW + p] = src[i];
}

// ---------------------------------------------------------------------------
// Driver
// ---------------------------------------------------------------------------
extern "C" void kernel_launch(void* const* d_in, const int* in_sizes, int n_in,
                              void* d_out, int out_size, void* d_ws, size_t ws_size,
                              hipStream_t stream)
{
    (void)in_sizes; (void)out_size; (void)ws_size;
    // Parameter flattening (setup_inputs dict insertion order):
    // 0 im_tar, 1 im_ref, enc_a: 2..25, enc_b: 26..49,
    // dec6: 50..61, dec5: 62..73, dec4: 74..85, dec3: 86..97, dec2: 98..109
    const float* P[128];
    for (int i = 0; i < n_in && i < 128; ++i) P[i] = (const float*)d_in[i];

    static const int ENCC[6] = {16, 32, 64, 96, 128, 192};
    static const int HL[7]   = {256, 128, 64, 32, 16, 8, 4};
    static const int WL[7]   = {448, 224, 112, 56, 28, 14, 7};

    float* base = (float*)d_ws;
    size_t off = 0;
    auto alloc = [&](size_t nf) { float* p = base + off; off += (nf + 63) & ~(size_t)63; return p; };

    float* tgt = alloc(2 * 3 * 256 * 448);
    float* ref = alloc(2 * 3 * 256 * 448);
    float* featA[6]; float* featB[6];
    for (int i = 0; i < 6; ++i) featA[i] = alloc((size_t)2 * ENCC[i] * HL[i + 1] * WL[i + 1]);
    for (int i = 0; i < 6; ++i) featB[i] = alloc((size_t)2 * ENCC[i] * HL[i + 1] * WL[i + 1]);
    float* s1     = alloc(2 * 128 * 64 * 112);
    float* s2     = alloc(2 * 128 * 64 * 112);
    float* concat = alloc(2 * 115 * 64 * 112);
    float* warped = alloc(2 * 32 * 64 * 112);
    float* flowb  = alloc(2 * 2 * 64 * 112);
    float* flowup = alloc(2 * 2 * 128 * 224);
    _Float16* whBuf = (_Float16*)alloc(192 * 3072 / 2 + 64);         // max padded weights (f16)
    _Float16* actH  = (_Float16*)alloc(2 * 128 * 64 * 112 / 2 + 64); // max activation (f16)
    _Float16* zpage = (_Float16*)alloc(64);                          // 128 zero halfs

    auto ewg = [](int total) { return dim3((unsigned)((total + 255) / 256)); };

    zfill_kernel<<<1, 128, 0, stream>>>(zpage, 128);

    auto conv = [&](const float* in, const float* wt, const float* bs, float* out,
                    int Cin, int Hin, int Win, int Cout, int stride, int relu) {
        int Hout = (Hin - 1) / stride + 1;
        int Wout = (Win - 1) / stride + 1;
        int Kp = ((Cin * 16 + 31) / 32) * 32;
        int wtot = Cout * Kp;
        wcvt_kernel<<<ewg(wtot), 256, 0, stream>>>(wt, whBuf, Cin, Kp, wtot);
        int atot = 2 * Cin * Hin * Win;
        acvt_kernel<<<ewg(atot), 256, 0, stream>>>(in, actH, atot);
        int WT = (Wout + 15) / 16;
        unsigned gx = (unsigned)(2 * Hout * WT);
        int n4 = Cout >> 6;
        if (n4 > 0) {
            conv3x3_wmma_f16<4><<<dim3(gx, (unsigned)n4), 32, 0, stream>>>(
                actH, whBuf, bs, zpage, out, Cin, Hin, Win, Cout, Hout, Wout, stride, relu, Kp, 0);
        }
        int rem = Cout & 63;
        if (rem > 0) {
            int mb = n4 << 6;
            int mtr = (rem + 15) >> 4;
            dim3 g(gx, 1);
            switch (mtr) {
            case 1: conv3x3_wmma_f16<1><<<g, 32, 0, stream>>>(
                        actH, whBuf, bs, zpage, out, Cin, Hin, Win, Cout, Hout, Wout, stride, relu, Kp, mb); break;
            case 2: conv3x3_wmma_f16<2><<<g, 32, 0, stream>>>(
                        actH, whBuf, bs, zpage, out, Cin, Hin, Win, Cout, Hout, Wout, stride, relu, Kp, mb); break;
            default: conv3x3_wmma_f16<3><<<g, 32, 0, stream>>>(
                        actH, whBuf, bs, zpage, out, Cin, Hin, Win, Cout, Hout, Wout, stride, relu, Kp, mb); break;
            }
        }
    };

    // ---- preprocess ----
    {
        int total = 2 * 3 * 256 * 448;
        submean_kernel<<<ewg(total), 256, 0, stream>>>(P[0], tgt, 256 * 448, total,
                                                       0.411618f, 0.434631f, 0.454253f);
        submean_kernel<<<ewg(total), 256, 0, stream>>>(P[1], ref, 256 * 448, total,
                                                       0.410782f, 0.433645f, 0.452793f);
    }

    // ---- encoders ----
    auto encode = [&](const float* img, int pbase, float** feat) {
        const float* x = img; int cin = 3;
        for (int i = 0; i < 6; ++i) {
            int cout = ENCC[i];
            conv(x, P[pbase + 4 * i], P[pbase + 4 * i + 1], s1,
                 cin, HL[i], WL[i], cout, 2, 1);
            conv(s1, P[pbase + 4 * i + 2], P[pbase + 4 * i + 3], feat[i],
                 cout, HL[i + 1], WL[i + 1], cout, 1, 1);
            x = feat[i]; cin = cout;
        }
    };
    encode(tgt, 2, featA);
    encode(ref, 26, featB);

    // ---- decoder chain ----
    auto runDec = [&](int lvl, const float* din, int nin, int Hc, int Wc, float* flowOut) {
        int pbase = 50 + (6 - lvl) * 12;
        int ch[7] = {nin, 128, 128, 96, 64, 32, 2};
        const float* cur = din;
        for (int i = 0; i < 6; ++i) {
            float* o = (i == 5) ? flowOut : ((i & 1) ? s2 : s1);
            conv(cur, P[pbase + 2 * i], P[pbase + 2 * i + 1], o,
                 ch[i], Hc, Wc, ch[i + 1], 1, (i < 5) ? 1 : 0);
            cur = o;
        }
    };

    // ---- level 6 ----
    {
        int Hc = HL[6], Wc = WL[6], C = ENCC[5];
        int total = 2 * 81 * Hc * Wc;
        corr_kernel<<<ewg(total), 256, 0, stream>>>(featA[5], featB[5], concat,
                                                    C, Hc, Wc, 81, 1.f / (float)C, total);
        runDec(6, concat, 81, Hc, Wc, flowb);
    }

    // ---- levels 5..2 ----
    for (int lvl = 5; lvl >= 2; --lvl) {
        int Hc = HL[lvl], Wc = WL[lvl];
        int Cf = ENCC[lvl - 1];
        int nin = 81 + Cf + 2;
        float wscale = 20.f / (float)(1 << lvl);

        int tflow = 2 * 2 * Hc * Wc;
        upsample2_kernel<<<ewg(tflow), 256, 0, stream>>>(flowb, flowup, 2,
                                                         HL[lvl + 1], WL[lvl + 1], 1.f, tflow);
        int tfeat = 2 * Cf * Hc * Wc;
        warp_kernel<<<ewg(tfeat), 256, 0, stream>>>(featB[lvl - 1], flowup, wscale,
                                                    warped, Cf, Hc, Wc, tfeat);
        int tcorr = 2 * 81 * Hc * Wc;
        corr_kernel<<<ewg(tcorr), 256, 0, stream>>>(featA[lvl - 1], warped, concat,
                                                    Cf, Hc, Wc, nin, 1.f / (float)Cf, tcorr);
        copycat_kernel<<<ewg(tfeat), 256, 0, stream>>>(featA[lvl - 1], concat,
                                                       Cf, Hc * Wc, nin, 81, tfeat);
        copycat_kernel<<<ewg(tflow), 256, 0, stream>>>(flowup, concat,
                                                       2, Hc * Wc, nin, 81 + Cf, tflow);
        runDec(lvl, concat, nin, Hc, Wc, flowb);
    }

    // ---- final upsampling: flow2 (64x112) -> 128x224 -> 256x448 * 20 ----
    {
        int t1 = 2 * 2 * 128 * 224;
        upsample2_kernel<<<ewg(t1), 256, 0, stream>>>(flowb, flowup, 2, 64, 112, 1.f, t1);
        int t2 = 2 * 2 * 256 * 448;
        upsample2_kernel<<<ewg(t2), 256, 0, stream>>>(flowup, (float*)d_out, 2, 128, 224,
                                                      20.f, t2);
    }
}
